// LSTMGraphPredictor_61057255080262
// MI455X (gfx1250) — matrix operand order repaired and checked by reading, not compiled, not run on hardware
//
#include <hip/hip_runtime.h>
#include <hip/hip_bf16.h>
#include <stdint.h>

// ---------------- problem constants ----------------
#define E_    256
#define H_    1500
#define HP_   1504          // H padded to mult of 32
#define G4_   6016          // 4 gate blocks of HP_ (mult of 64)
#define B_    32
#define N_    32
#define P_    496           // N*(N-1)/2
#define MBP_  15872         // B_*P_
#define QIN_  512
#define Q1_   600
#define Q1P_  640           // padded to mult of 64 (also next layer's Kpad, mult of 32)
#define Q_    300
#define QP_   320
#define EV1_  1000
#define EV1P_ 1024
#define EV2_  100
#define EV2P_ 128           // padded to mult of 64

typedef __bf16 v8bf  __attribute__((ext_vector_type(8)));
typedef __bf16 v16bf __attribute__((ext_vector_type(16)));
typedef float  v8f   __attribute__((ext_vector_type(8)));

__device__ __forceinline__ unsigned short f2bf(float f) {
  union { float f; unsigned int u; } v; v.f = f;
  unsigned int u = v.u;
  unsigned int r = u + 0x7FFFu + ((u >> 16) & 1u);       // RNE
  if ((u & 0x7F800000u) == 0x7F800000u) r = u;           // inf/nan passthrough
  return (unsigned short)(r >> 16);
}

// two contiguous 16-byte chunks -> one 16-element bf16 fragment
__device__ __forceinline__ v16bf ldfrag(const unsigned short* p0, const unsigned short* p1) {
  v8bf lo = *reinterpret_cast<const v8bf*>(p0);
  v8bf hi = *reinterpret_cast<const v8bf*>(p1);
  return __builtin_shufflevector(lo, hi, 0,1,2,3,4,5,6,7,8,9,10,11,12,13,14,15);
}

// ---------------- elementwise / repack kernels ----------------

// per-(b,e) column sort of x along parts axis; writes seq bf16 [N_][B_][E_]
__global__ void k_sort_seq(const float* __restrict__ x, unsigned short* __restrict__ seq) {
  int t = blockIdx.x * blockDim.x + threadIdx.x;
  if (t >= B_ * E_) return;
  int b = t / E_, e = t % E_;
  float v[N_];
  const float* src = x + (size_t)b * N_ * E_ + e;
  for (int n = 0; n < N_; ++n) v[n] = src[(size_t)n * E_];
  for (int i = 1; i < N_; ++i) { float key = v[i]; int j = i - 1;
    while (j >= 0 && v[j] > key) { v[j+1] = v[j]; --j; } v[j+1] = key; }
  for (int n = 0; n < N_; ++n) seq[((size_t)n * B_ + b) * E_ + e] = f2bf(v[n]);
}

// transpose+convert gate weights: dst[Kpad][G4_]; col c -> (gate=c/HP_, h=c%HP_)
__global__ void k_gate_wt(unsigned short* __restrict__ dst, const float* __restrict__ src,
                          int Ksrc, int Kpad) {
  size_t t = (size_t)blockIdx.x * blockDim.x + threadIdx.x;
  size_t tot = (size_t)Kpad * G4_;
  if (t >= tot) return;
  int k = (int)(t / G4_), c = (int)(t % G4_);
  int gate = c / HP_, h = c % HP_;
  unsigned short o = 0;
  if (h < H_ && k < Ksrc) o = f2bf(src[(size_t)(gate * H_ + h) * Ksrc + k]);
  dst[t] = o;
}

// combined gate bias (b_ih + b_hh), gate-block padded layout [G4_]
__global__ void k_gate_bias(float* __restrict__ gb, const float* __restrict__ bih,
                            const float* __restrict__ bhh) {
  int c = blockIdx.x * blockDim.x + threadIdx.x;
  if (c >= G4_) return;
  int gate = c / HP_, h = c % HP_;
  gb[c] = (h < H_) ? (bih[gate * H_ + h] + bhh[gate * H_ + h]) : 0.f;
}

// generic transpose+convert: dst[Kpad][Npad] bf16 from src rows [Nsrc][srcStride]
__global__ void k_tr_cvt(unsigned short* __restrict__ dst, const float* __restrict__ src,
                         int Ksrc, int Kpad, int Nsrc, int Npad, int srcStride, int srcCol0) {
  size_t t = (size_t)blockIdx.x * blockDim.x + threadIdx.x;
  size_t tot = (size_t)Kpad * Npad;
  if (t >= tot) return;
  int k = (int)(t / Npad), c = (int)(t % Npad);
  unsigned short o = 0;
  if (k < Ksrc && c < Nsrc) o = f2bf(src[(size_t)c * srcStride + srcCol0 + k]);
  dst[t] = o;
}

__global__ void k_zero(unsigned int* __restrict__ p, int n) {
  int t = blockIdx.x * blockDim.x + threadIdx.x;
  if (t < n) p[t] = 0u;
}

// triu(k=1) pair index table, row-major order
__global__ void k_pairidx(int* __restrict__ pidx) {
  int p = blockIdx.x * blockDim.x + threadIdx.x;
  if (p >= P_) return;
  int i = 0, base = 0;
  while (p >= base + (N_ - 1 - i)) { base += (N_ - 1 - i); ++i; }
  int j = i + 1 + (p - base);
  pidx[2 * p] = i; pidx[2 * p + 1] = j;
}

// pairs_bf[b][p][512] = bf16( concat(x[b,i],x[b,j]) + PE(pos0) ); PE adds 1 on odd dims
__global__ void k_build_pairs(const float* __restrict__ x, const int* __restrict__ pidx,
                              unsigned short* __restrict__ pairs) {
  size_t t = (size_t)blockIdx.x * blockDim.x + threadIdx.x;
  size_t tot = (size_t)MBP_ * (2 * E_);
  if (t >= tot) return;
  int d = (int)(t & 511);
  int m = (int)(t >> 9);
  int b = m / P_, p = m % P_;
  int part = (d < E_) ? pidx[2 * p] : pidx[2 * p + 1];
  int e = d & (E_ - 1);
  float val = x[((size_t)b * N_ + part) * E_ + e] + ((d & 1) ? 1.0f : 0.0f);
  pairs[t] = f2bf(val);
}

// ---------------- WMMA GEMM: out[m,n] = act( A_bf[m,:]·Wt_bf[:,n] + bias + rowadd ) ----------
// A: [M][Kpad] bf16 row-major, Wt: [Kpad][Npad] bf16 (i.e. W transposed).
// M % 32 == 0, Kpad % 32 == 0, Npad % 64 == 0 (all loops static -> accs stay in VGPRs).
__global__ void k_gemm_bf16(const unsigned short* __restrict__ A,
                            const unsigned short* __restrict__ Wt,
                            const float* __restrict__ bias,
                            float* __restrict__ outF, int ldF,
                            unsigned short* __restrict__ outB,
                            const float* __restrict__ rowadd, int raStride, int raDiv,
                            int M, int Kpad, int Npad, int Nout, int doRelu, int totalWaves) {
  int wid  = (int)((blockIdx.x * blockDim.x + threadIdx.x) >> 5);
  int lane = threadIdx.x & 31;
  if (wid >= totalWaves) return;
  int ntiles = Npad >> 6;
  int n0 = (wid % ntiles) << 6;
  int m0 = (wid / ntiles) << 5;

  const v8f vz = {0.f,0.f,0.f,0.f,0.f,0.f,0.f,0.f};
  v8f acc[2][4];
#pragma unroll
  for (int mt = 0; mt < 2; ++mt)
#pragma unroll
    for (int st = 0; st < 4; ++st) acc[mt][st] = vz;

  int kb  = (lane >> 4) << 3;   // A-frag K sub-offset: lanes 0-15 -> 0, 16-31 -> 8
  int mlo = lane & 15;
  const unsigned short* ar0 = A + (size_t)(m0 + mlo) * Kpad;
  const unsigned short* ar1 = A + (size_t)(m0 + 16 + mlo) * Kpad;

  for (int k0 = 0; k0 < Kpad; k0 += 32) {
    v16bf a0 = ldfrag(ar0 + k0 + kb, ar0 + k0 + kb + 16);
    v16bf a1 = ldfrag(ar1 + k0 + kb, ar1 + k0 + kb + 16);
    const unsigned short* brow = Wt + (size_t)(k0 + kb) * Npad + n0 + mlo;  // lane%16 == N, elements over K (table order)
    v16bf bfr[4];
#pragma unroll
    for (int st = 0; st < 4; ++st)
#pragma unroll
      for (int s = 0; s < 16; ++s) bfr[st][s] = __builtin_bit_cast(__bf16, brow[(size_t)(s < 8 ? s : s + 8) * Npad + st * 16]);
#pragma unroll
    for (int st = 0; st < 4; ++st) {
      acc[0][st] = __builtin_amdgcn_wmma_f32_16x16x32_bf16(false, a0, false, bfr[st], (short)0, acc[0][st], false, false);
      acc[1][st] = __builtin_amdgcn_wmma_f32_16x16x32_bf16(false, a1, false, bfr[st], (short)0, acc[1][st], false, false);
    }
  }

  int mh = (lane >> 4) << 3;    // C/D layout: lanes 16-31 hold M = r+8
#pragma unroll
  for (int mt = 0; mt < 2; ++mt) {
#pragma unroll
    for (int st = 0; st < 4; ++st) {
      int nn = n0 + st * 16 + mlo;
      float bv = (bias && nn < Nout) ? bias[nn] : 0.f;
#pragma unroll
      for (int r = 0; r < 8; ++r) {
        int mm = m0 + mt * 16 + mh + r;
        float val = acc[mt][st][r] + bv;
        if (rowadd && nn < Nout) val += rowadd[(size_t)(mm / raDiv) * raStride + nn];
        if (doRelu) val = fmaxf(val, 0.f);
        if (outF) { if (nn < Nout) outF[(size_t)mm * ldF + nn] = val; }
        if (outB) outB[(size_t)mm * Npad + nn] = (nn < Nout) ? f2bf(val) : (unsigned short)0;
      }
    }
  }
}

// ---------------- LSTM recurrence step ----------------
// g = Gx[step] + h_src @ W_hh^T (gate-blocked), then gate math; h double-buffered.
// One wave owns one 16-wide h tile for all 32 batches and all 4 gates.
__global__ void k_lstm_step(const unsigned short* __restrict__ hsrc,
                            unsigned short* __restrict__ hdst,
                            float* __restrict__ cs,
                            const float* __restrict__ gx,          // + step*B_*G4_
                            const unsigned short* __restrict__ wt) // [HP_][G4_]
{
  int wid  = (int)((blockIdx.x * blockDim.x + threadIdx.x) >> 5);
  int lane = threadIdx.x & 31;
  if (wid >= HP_ / 16) return;           // 94 tiles
  int h0 = wid * 16;

  const v8f vz = {0.f,0.f,0.f,0.f,0.f,0.f,0.f,0.f};
  v8f acc[4][2];
#pragma unroll
  for (int g = 0; g < 4; ++g)
#pragma unroll
    for (int mt = 0; mt < 2; ++mt) acc[g][mt] = vz;

  int kb  = (lane >> 4) << 3;
  int mlo = lane & 15;
  const unsigned short* ar0 = hsrc + (size_t)mlo * HP_;
  const unsigned short* ar1 = hsrc + (size_t)(16 + mlo) * HP_;

  for (int k0 = 0; k0 < HP_; k0 += 32) {
    v16bf a0 = ldfrag(ar0 + k0 + kb, ar0 + k0 + kb + 16);
    v16bf a1 = ldfrag(ar1 + k0 + kb, ar1 + k0 + kb + 16);
    const unsigned short* brow = wt + (size_t)(k0 + kb) * G4_ + h0 + mlo;
    v16bf bfr[4];
#pragma unroll
    for (int g = 0; g < 4; ++g)
#pragma unroll
      for (int s = 0; s < 16; ++s) bfr[g][s] = __builtin_bit_cast(__bf16, brow[(size_t)(s < 8 ? s : s + 8) * G4_ + g * HP_]);
#pragma unroll
    for (int g = 0; g < 4; ++g) {
      acc[g][0] = __builtin_amdgcn_wmma_f32_16x16x32_bf16(false, a0, false, bfr[g], (short)0, acc[g][0], false, false);
      acc[g][1] = __builtin_amdgcn_wmma_f32_16x16x32_bf16(false, a1, false, bfr[g], (short)0, acc[g][1], false, false);
    }
  }

  int mh = (lane >> 4) << 3;
  int hh = h0 + mlo;
#pragma unroll
  for (int mt = 0; mt < 2; ++mt) {
#pragma unroll
    for (int r = 0; r < 8; ++r) {
      int b = mt * 16 + mh + r;
      const float* gxb = gx + (size_t)b * G4_;
      float gi = acc[0][mt][r] + gxb[0 * HP_ + hh];
      float gf = acc[1][mt][r] + gxb[1 * HP_ + hh];
      float gg = acc[2][mt][r] + gxb[2 * HP_ + hh];
      float go = acc[3][mt][r] + gxb[3 * HP_ + hh];
      float si = 1.f / (1.f + __expf(-gi));
      float sf = 1.f / (1.f + __expf(-gf));
      float so = 1.f / (1.f + __expf(-go));
      float tg = tanhf(gg);
      size_t ci = (size_t)b * HP_ + hh;
      float c  = sf * cs[ci] + si * tg;
      cs[ci]   = c;
      hdst[ci] = f2bf(so * tanhf(c));
    }
  }
}

// ---------------- tiny e3/e4 tail (VALU) ----------------
__global__ void k_e34(const float* __restrict__ a, const float* __restrict__ w3,
                      const float* __restrict__ b3, const float* __restrict__ w4,
                      const float* __restrict__ b4, float* __restrict__ out) {
  int m = blockIdx.x * blockDim.x + threadIdx.x;
  if (m >= MBP_) return;
  const float* row = a + (size_t)m * EV2_;
  float h3[10];
#pragma unroll
  for (int o = 0; o < 10; ++o) {
    float s = b3[o];
    const float* w = w3 + o * EV2_;
    for (int k = 0; k < EV2_; ++k) s += w[k] * row[k];
    h3[o] = fmaxf(s, 0.f);
  }
  float s4 = b4[0];
#pragma unroll
  for (int o = 0; o < 10; ++o) s4 += w4[o] * h3[o];
  out[m] = fmaxf(s4, 0.f);
}

// ---------------- host orchestration ----------------
extern "C" void kernel_launch(void* const* d_in, const int* in_sizes, int n_in,
                              void* d_out, int out_size, void* d_ws, size_t ws_size,
                              hipStream_t stream) {
  (void)in_sizes; (void)n_in; (void)out_size; (void)ws_size;
  const float* x    = (const float*)d_in[0];
  const float* w_ih = (const float*)d_in[1];
  const float* w_hh = (const float*)d_in[2];
  const float* b_ih = (const float*)d_in[3];
  const float* b_hh = (const float*)d_in[4];
  const float* q1_w = (const float*)d_in[5];
  const float* q1_b = (const float*)d_in[6];
  const float* q2_w = (const float*)d_in[7];
  const float* q2_b = (const float*)d_in[8];
  const float* e1_w = (const float*)d_in[9];
  const float* e1_b = (const float*)d_in[10];
  const float* e2_w = (const float*)d_in[11];
  const float* e2_b = (const float*)d_in[12];
  const float* e3_w = (const float*)d_in[13];
  const float* e3_b = (const float*)d_in[14];
  const float* e4_w = (const float*)d_in[15];
  const float* e4_b = (const float*)d_in[16];
  float* out = (float*)d_out;

  char* ws = (char*)d_ws;
  size_t off = 0;
  auto alloc = [&](size_t bytes) -> char* {
    char* p = ws + off;
    off = (off + bytes + 255) & ~(size_t)255;
    return p;
  };
  unsigned short* seq    = (unsigned short*)alloc((size_t)N_ * B_ * E_ * 2);
  unsigned short* wt_ih  = (unsigned short*)alloc((size_t)E_ * G4_ * 2);
  unsigned short* wt_hh  = (unsigned short*)alloc((size_t)HP_ * G4_ * 2);
  float*          gb     = (float*)        alloc((size_t)G4_ * 4);
  float*          Gx     = (float*)        alloc((size_t)N_ * B_ * G4_ * 4);
  unsigned short* hbuf   = (unsigned short*)alloc((size_t)2 * B_ * HP_ * 2);
  float*          cs     = (float*)        alloc((size_t)B_ * HP_ * 4);
  float*          planC  = (float*)        alloc((size_t)B_ * EV1_ * 4);
  int*            pidx   = (int*)          alloc((size_t)P_ * 2 * 4);
  unsigned short* pairs  = (unsigned short*)alloc((size_t)MBP_ * 2 * E_ * 2);
  unsigned short* q1_wt  = (unsigned short*)alloc((size_t)QIN_ * Q1P_ * 2);
  unsigned short* q1o    = (unsigned short*)alloc((size_t)MBP_ * Q1P_ * 2);
  unsigned short* q2_wt  = (unsigned short*)alloc((size_t)Q1P_ * QP_ * 2);
  unsigned short* q2o    = (unsigned short*)alloc((size_t)MBP_ * QP_ * 2);
  unsigned short* e1h_t  = (unsigned short*)alloc((size_t)HP_ * EV1P_ * 2);
  unsigned short* e1q_t  = (unsigned short*)alloc((size_t)QP_ * EV1P_ * 2);
  unsigned short* e1o    = (unsigned short*)alloc((size_t)MBP_ * EV1P_ * 2);
  unsigned short* e2_t   = (unsigned short*)alloc((size_t)EV1P_ * EV2P_ * 2);
  float*          e2o    = (float*)        alloc((size_t)MBP_ * EV2_ * 4);

  auto blks = [](size_t n, int t) { return (unsigned)((n + t - 1) / t); };
  auto gemm_blocks = [](int waves) { return (unsigned)((waves * 32 + 127) / 128); };

  // --- stage 0: sort + repack weights + tables ---
  k_sort_seq<<<blks((size_t)B_ * E_, 256), 256, 0, stream>>>(x, seq);
  k_gate_wt<<<blks((size_t)E_ * G4_, 256), 256, 0, stream>>>(wt_ih, w_ih, E_, E_);
  k_gate_wt<<<blks((size_t)HP_ * G4_, 256), 256, 0, stream>>>(wt_hh, w_hh, H_, HP_);
  k_gate_bias<<<blks(G4_, 256), 256, 0, stream>>>(gb, b_ih, b_hh);
  k_tr_cvt<<<blks((size_t)QIN_ * Q1P_, 256), 256, 0, stream>>>(q1_wt, q1_w, QIN_, QIN_, Q1_, Q1P_, QIN_, 0);
  k_tr_cvt<<<blks((size_t)Q1P_ * QP_, 256), 256, 0, stream>>>(q2_wt, q2_w, Q1_, Q1P_, Q_, QP_, Q1_, 0);
  k_tr_cvt<<<blks((size_t)HP_ * EV1P_, 256), 256, 0, stream>>>(e1h_t, e1_w, H_, HP_, EV1_, EV1P_, H_ + Q_, 0);
  k_tr_cvt<<<blks((size_t)QP_ * EV1P_, 256), 256, 0, stream>>>(e1q_t, e1_w, Q_, QP_, EV1_, EV1P_, H_ + Q_, H_);
  k_tr_cvt<<<blks((size_t)EV1P_ * EV2P_, 256), 256, 0, stream>>>(e2_t, e2_w, EV1_, EV1P_, EV2_, EV2P_, EV1_, 0);
  k_pairidx<<<blks(P_, 256), 256, 0, stream>>>(pidx);
  k_zero<<<blks((size_t)2 * B_ * HP_ * 2 / 4, 256), 256, 0, stream>>>((unsigned int*)hbuf, 2 * B_ * HP_ * 2 / 4);
  k_zero<<<blks((size_t)B_ * HP_, 256), 256, 0, stream>>>((unsigned int*)cs, B_ * HP_);

  // --- stage 1: Gx = seq @ W_ih^T + (b_ih+b_hh)  as one [1024 x 6016 x 256] GEMM ---
  {
    int waves = (N_ * B_ / 32) * (G4_ / 64); // 32 * 94
    k_gemm_bf16<<<gemm_blocks(waves), 128, 0, stream>>>(
        seq, wt_ih, gb, Gx, G4_, nullptr, nullptr, 0, 1,
        N_ * B_, E_, G4_, G4_, 0, waves);
  }

  // --- stage 2: 32 sequential LSTM steps (h double-buffered) ---
  for (int s = 0; s < N_; ++s) {
    unsigned short* hsrc = hbuf + (size_t)(s & 1) * B_ * HP_;
    unsigned short* hdst = hbuf + (size_t)((s + 1) & 1) * B_ * HP_;
    k_lstm_step<<<gemm_blocks(HP_ / 16), 128, 0, stream>>>(
        hsrc, hdst, cs, Gx + (size_t)s * B_ * G4_, wt_hh);
  }
  unsigned short* plan = hbuf; // after 32 steps final h is in buffer 0

  // --- stage 3: planC = plan @ e1_w[:, :H]^T + e1_b   [32 x 1000] ---
  {
    int waves = 1 * (EV1P_ / 64);
    k_gemm_bf16<<<gemm_blocks(waves), 128, 0, stream>>>(
        plan, e1h_t, e1_b, planC, EV1_, nullptr, nullptr, 0, 1,
        B_, HP_, EV1P_, EV1_, 0, waves);
  }

  // --- stage 4: pair pipeline ---
  k_build_pairs<<<blks((size_t)MBP_ * 2 * E_, 256), 256, 0, stream>>>(x, pidx, pairs);
  {
    int waves = (MBP_ / 32) * (Q1P_ / 64);
    k_gemm_bf16<<<gemm_blocks(waves), 128, 0, stream>>>(
        pairs, q1_wt, q1_b, nullptr, 0, q1o, nullptr, 0, 1,
        MBP_, QIN_, Q1P_, Q1_, 1, waves);
  }
  {
    int waves = (MBP_ / 32) * (QP_ / 64);
    k_gemm_bf16<<<gemm_blocks(waves), 128, 0, stream>>>(
        q1o, q2_wt, q2_b, nullptr, 0, q2o, nullptr, 0, 1,
        MBP_, Q1P_, QP_, Q_, 1, waves);
  }
  {
    int waves = (MBP_ / 32) * (EV1P_ / 64);
    k_gemm_bf16<<<gemm_blocks(waves), 128, 0, stream>>>(
        q2o, e1q_t, nullptr, nullptr, 0, e1o, planC, EV1_, P_,
        MBP_, QP_, EV1P_, EV1_, 1, waves);
  }
  {
    int waves = (MBP_ / 32) * (EV2P_ / 64);
    k_gemm_bf16<<<gemm_blocks(waves), 128, 0, stream>>>(
        e1o, e2_t, e2_b, e2o, EV2_, nullptr, nullptr, 0, 1,
        MBP_, EV1P_, EV2P_, EV2_, 1, waves);
  }

  // --- stage 5: e3/e4 tail ---
  k_e34<<<blks(MBP_, 256), 256, 0, stream>>>(e2o, e3_w, e3_b, e4_w, e4_b, out);
}
